// SparseMultiHeadAttention_54219667145587
// MI455X (gfx1250) — matrix operand order repaired and checked by reading, not compiled, not run on hardware
//
#include <hip/hip_runtime.h>

typedef unsigned int u32;
typedef unsigned short u16;
typedef __attribute__((ext_vector_type(4))) u32 v4u;
typedef __attribute__((ext_vector_type(16))) __bf16 v16bf;
typedef __attribute__((ext_vector_type(8))) float v8f;

// ---------- helpers ----------
__device__ __forceinline__ u16 f2bf(float x) {          // fp32 -> bf16 RNE
  u32 u = __float_as_uint(x);
  u += 0x7fffu + ((u >> 16) & 1u);
  return (u16)(u >> 16);
}

union Frag { v4u u[2]; v16bf v; };

// 16-bit A-matrix 16x32 layout (ISA 7.12.2): lane row-major, two 16B chunks 32B apart
__device__ __forceinline__ v16bf frag_strided(const void* p) {
  Frag f;
  f.u[0] = *(const v4u*)((const char*)p);
  f.u[1] = *(const v4u*)((const char*)p + 32);
  return f.v;
}
// pre-swizzled B fragments: lane's 16 bf16 contiguous (32 bytes)
__device__ __forceinline__ v16bf frag_linear(const void* p) {
  Frag f;
  f.u[0] = *(const v4u*)((const char*)p);
  f.u[1] = *(const v4u*)((const char*)p + 16);
  return f.v;
}
__device__ __forceinline__ v8f wmma_bf16(v16bf a, v16bf b, v8f c) {
  return __builtin_amdgcn_wmma_f32_16x16x32_bf16(false, a, false, b, (short)0, c,
                                                 false, false);
}

#if defined(__HIP_DEVICE_COMPILE__) && \
    __has_builtin(__builtin_amdgcn_global_load_async_to_lds_b128) && \
    __has_builtin(__builtin_amdgcn_s_wait_asynccnt)
#define CDNA5_ASYNC 1
typedef int v4i_raw __attribute__((vector_size(16)));
typedef __attribute__((address_space(1))) v4i_raw* as1_v4i_p;
typedef __attribute__((address_space(3))) v4i_raw* as3_v4i_p;
__device__ __forceinline__ void async_cp16(const void* g, void* l) {
  __builtin_amdgcn_global_load_async_to_lds_b128(
      (as1_v4i_p)(void*)g, (as3_v4i_p)l, 0, 0);
}
__device__ __forceinline__ void async_wait0() {
  __builtin_amdgcn_s_wait_asynccnt(0);
}
#else
#define CDNA5_ASYNC 0
#endif

// ---------- K0: pack weights into B-fragment-linear bf16 ----------
// dest flat = ((nt*16 + k)*32 + lane)*16 + e ; K = k*32 + (e&7) + (lane>=16?8:0) + (e>=8?16:0),
// N = nt*16 + (lane&15)
__global__ __launch_bounds__(256) void k_pack_w(const float* __restrict__ wqkv,
                                                const float* __restrict__ wout,
                                                u16* __restrict__ wq_p,
                                                u16* __restrict__ wo_p) {
  long i = (long)blockIdx.x * 256 + threadIdx.x;
  if (i < 786432) {
    int e = i & 15, lane = (i >> 4) & 31, k = (i >> 9) & 15, nt = (int)(i >> 13);
    int K = k * 32 + (e & 7) + ((lane >= 16) ? 8 : 0) + ((e >= 8) ? 16 : 0);
    int N = nt * 16 + (lane & 15);
    wq_p[i] = f2bf(wqkv[(size_t)K * 1536 + N]);
  } else {
    long f = i - 786432;
    int e = f & 15, lane = (f >> 4) & 31, k = (f >> 9) & 15, nt = (int)(f >> 13);
    int K = k * 32 + (e & 7) + ((lane >= 16) ? 8 : 0) + ((e >= 8) ? 16 : 0);
    int N = nt * 16 + (lane & 15);
    wo_p[f] = f2bf(wout[(size_t)K * 512 + N]);
  }
}

// ---------- K1: QKV GEMM (16 rows/block, all 1536 cols) ----------
// A-slab (16 k-steps) preloaded once into 128 VGPRs, reused for 12 col tiles.
__global__ __launch_bounds__(256) void k_qkv_gemm(const float* __restrict__ x,
                                                  const u16* __restrict__ wq_p,
                                                  const float* __restrict__ b_qkv,
                                                  float* __restrict__ qkv) {
  __shared__ __align__(16) u16 xs[16 * 512];       // 16 KB
  int t = threadIdx.x;
  int row0 = blockIdx.x * 16;
  for (int i = t; i < 16 * 512; i += 256)
    xs[i] = f2bf(x[(size_t)row0 * 512 + i]);
  __syncthreads();

  int wave = t >> 5, lane = t & 31;
  int mr = lane & 15, hi = lane >> 4;
  const char* abase = (const char*)xs + mr * 1024 + hi * 16;
  v16bf afrag[16];
#pragma unroll
  for (int k = 0; k < 16; ++k) afrag[k] = frag_strided(abase + k * 64);

  for (int i = 0; i < 12; ++i) {
    int ti = wave * 12 + i;                         // col tile 0..95
    v8f acc = {};
    const u16* bp = wq_p + (size_t)ti * 8192 + lane * 16;
#pragma unroll
    for (int k = 0; k < 16; ++k)
      acc = wmma_bf16(afrag[k], frag_linear(bp + k * 512), acc);
    int col = ti * 16 + mr;                         // C layout: N = lane&15
    float bias = b_qkv[col];
#pragma unroll
    for (int r = 0; r < 8; ++r)
      qkv[(size_t)(row0 + r + 8 * hi) * 1536 + col] = acc[r] + bias;
  }
}

// ---------- K2: per-head RMS norm + analytic window gather/pack ----------
__global__ __launch_bounds__(256) void k_rms_scatter(const float* __restrict__ qkv,
                                                     const float* __restrict__ gq,
                                                     const float* __restrict__ gk,
                                                     u16* __restrict__ q_g,
                                                     u16* __restrict__ k_p,
                                                     u16* __restrict__ v_p) {
  int t = threadIdx.x;
  int row = blockIdx.x * 16 + (t >> 4);
  int slot = t & 15;                                 // 0..7 q heads, 8..15 k heads
  int xx = row & 31, y = (row >> 5) & 31, z = row >> 10;
  int w = ((xx >> 3) * 4 + (y >> 3)) * 4 + (z >> 3);
  int m = (((z & 7) * 8 + (y & 7)) * 8 + (xx & 7));

  int head = slot & 7;
  bool isK = slot >= 8;
  const float* src = qkv + (size_t)row * 1536 + slot * 64;
  float vals[64];
  float ss = 0.f;
#pragma unroll
  for (int d = 0; d < 64; ++d) { vals[d] = src[d]; ss += vals[d] * vals[d]; }
  float nrm = fmaxf(sqrtf(ss), 1e-12f);
  float sc = 8.0f / nrm;                             // sqrt(D)=8
  const float* gamma = (isK ? gk : gq) + head * 64;

  if (!isK) {                                        // Q: row-major [w][h][m][d]
    u16* dst = q_g + (((size_t)(w * 8 + head)) * 512 + m) * 64;
#pragma unroll
    for (int d = 0; d < 64; ++d) dst[d] = f2bf(vals[d] * sc * gamma[d]);
  } else {                                           // K^T as B-frag pack (K-dim = d)
    size_t base = ((size_t)(w * 8 + head)) * 32768;
#pragma unroll
    for (int d = 0; d < 64; ++d) {
      int ti = (d >> 5) * 32 + (m >> 4);
      int lane = (m & 15) + (((d >> 3) & 1) ? 16 : 0);
      k_p[base + ((size_t)ti * 32 + lane) * 16 + (d & 7) + ((d & 16) ? 8 : 0)] =
          f2bf(vals[d] * sc * gamma[d]);
    }
  }
  // V as B-frag pack (K-dim = m). Each thread handles 32 of 512 v cols.
  const float* vsrc = qkv + (size_t)row * 1536 + 1024;
  int kstep = m >> 5, km = m & 31;
#pragma unroll
  for (int i = 0; i < 32; ++i) {
    int c = slot * 32 + i;
    int h2 = c >> 6, d = c & 63;
    size_t base = ((size_t)(w * 8 + h2)) * 32768;
    int ti = kstep * 4 + (d >> 4);
    int lane = (d & 15) + (((km >> 3) & 1) ? 16 : 0);
    v_p[base + ((size_t)ti * 32 + lane) * 16 + (km & 7) + ((km & 16) ? 8 : 0)] = f2bf(vsrc[c]);
  }
}

// ---------- K3: attention per (window, head, 128-row slab) ----------
__global__ __launch_bounds__(256) void k_attn(const u16* __restrict__ q_g,
                                              const u16* __restrict__ k_p,
                                              const u16* __restrict__ v_p,
                                              u16* __restrict__ h_g) {
  extern __shared__ __align__(16) char smem[];       // 16KB Q + 256KB scores
  u16* qs = (u16*)smem;
  char* sc = smem + 16384;                           // 128 rows, 2048B stride
  int blk = blockIdx.x;
  int wh = blk >> 2;                                 // w*8 + h
  int m0 = (blk & 3) * 128;
  int t = threadIdx.x, wave = t >> 5, lane = t & 31;
  int mr = lane & 15, hi = lane >> 4;

  const char* qsrc = (const char*)(q_g + ((size_t)wh * 512 + m0) * 64);
#if CDNA5_ASYNC
  for (int b = t * 16; b < 16384; b += 4096) async_cp16(qsrc + b, smem + b);
  async_wait0();
#else
  for (int b = t * 16; b < 16384; b += 4096)
    *(v4u*)(smem + b) = *(const v4u*)(qsrc + b);
#endif
  __syncthreads();

  // ---- scores = Q K^T / 8 ----
  const char* abase = (const char*)qs + (wave * 16 + mr) * 128 + hi * 16;
  v16bf qfrag0 = frag_strided(abase);
  v16bf qfrag1 = frag_strided(abase + 64);
  const u16* kb = k_p + (size_t)wh * 32768 + lane * 16;
  for (int nt = 0; nt < 32; ++nt) {
    v8f acc = {};
    acc = wmma_bf16(qfrag0, frag_linear(kb + (size_t)nt * 512), acc);
    acc = wmma_bf16(qfrag1, frag_linear(kb + (size_t)(32 + nt) * 512), acc);
#pragma unroll
    for (int r = 0; r < 8; ++r) {
      int mm = wave * 16 + r + 8 * hi;
      ((float*)(sc + (size_t)mm * 2048))[nt * 16 + mr] = acc[r] * 0.125f;
    }
  }
  __syncthreads();

  // ---- softmax: wave-parallel per row, probs overwritten in place as bf16 ----
  for (int rr = 0; rr < 16; ++rr) {
    int row = wave * 16 + rr;
    const float* srow = (const float*)(sc + (size_t)row * 2048);
    float v[16];
#pragma unroll
    for (int i = 0; i < 16; ++i) v[i] = srow[lane * 16 + i];
    float mx = v[0];
#pragma unroll
    for (int i = 1; i < 16; ++i) mx = fmaxf(mx, v[i]);
    for (int off = 16; off; off >>= 1) mx = fmaxf(mx, __shfl_xor(mx, off, 32));
    float sum = 0.f;
#pragma unroll
    for (int i = 0; i < 16; ++i) { v[i] = __expf(v[i] - mx); sum += v[i]; }
    for (int off = 16; off; off >>= 1) sum += __shfl_xor(sum, off, 32);
    float inv = 1.0f / sum;
    u32 p[8];
#pragma unroll
    for (int i = 0; i < 8; ++i)
      p[i] = (u32)f2bf(v[2 * i] * inv) | ((u32)f2bf(v[2 * i + 1] * inv) << 16);
    v4u* dst = (v4u*)((char*)sc + (size_t)row * 2048 + lane * 32);
    v4u d0 = {p[0], p[1], p[2], p[3]}, d1 = {p[4], p[5], p[6], p[7]};
    dst[0] = d0; dst[1] = d1;
  }
  __syncthreads();

  // ---- out = P V : preload prob A-slab (128 VGPRs), stream V ----
  const char* pbase = sc + (size_t)(wave * 16 + mr) * 2048 + hi * 16;
  v16bf pfrag[16];
#pragma unroll
  for (int k = 0; k < 16; ++k) pfrag[k] = frag_strided(pbase + k * 64);

  const u16* vb = v_p + (size_t)wh * 32768 + lane * 16;
  int head = wh & 7, w = wh >> 3;
  for (int dt = 0; dt < 4; ++dt) {
    v8f acc = {};
#pragma unroll
    for (int k = 0; k < 16; ++k)
      acc = wmma_bf16(pfrag[k], frag_linear(vb + (size_t)(k * 4 + dt) * 512), acc);
#pragma unroll
    for (int r = 0; r < 8; ++r) {
      int m = m0 + wave * 16 + r + 8 * hi;           // window-local index
      int z = (w & 3) * 8 + (m >> 6);
      int y = ((w >> 2) & 3) * 8 + ((m >> 3) & 7);
      int xx = (w >> 4) * 8 + (m & 7);
      int n = (z << 10) | (y << 5) | xx;             // inverse permutation
      h_g[(size_t)n * 512 + head * 64 + dt * 16 + mr] = f2bf(acc[r]);
    }
  }
}

// ---------- K4: output projection (16 rows/block) ----------
__global__ __launch_bounds__(256) void k_out(const u16* __restrict__ h_g,
                                             const u16* __restrict__ wo_p,
                                             const float* __restrict__ b_out,
                                             float* __restrict__ out) {
  __shared__ __align__(16) u16 hs[16 * 512];        // 16 KB
  int t = threadIdx.x;
  int row0 = blockIdx.x * 16;
  const char* hsrc = (const char*)(h_g + (size_t)row0 * 512);
#if CDNA5_ASYNC
  for (int b = t * 16; b < 16384; b += 4096) async_cp16(hsrc + b, (char*)hs + b);
  async_wait0();
#else
  for (int b = t * 16; b < 16384; b += 4096)
    *(v4u*)((char*)hs + b) = *(const v4u*)(hsrc + b);
#endif
  __syncthreads();

  int wave = t >> 5, lane = t & 31;
  int mr = lane & 15, hi = lane >> 4;
  const char* abase = (const char*)hs + mr * 1024 + hi * 16;
  v16bf afrag[16];
#pragma unroll
  for (int k = 0; k < 16; ++k) afrag[k] = frag_strided(abase + k * 64);

  for (int i = 0; i < 4; ++i) {
    int ti = wave * 4 + i;                           // col tile 0..31
    v8f acc = {};
    const u16* bp = wo_p + (size_t)ti * 8192 + lane * 16;
#pragma unroll
    for (int k = 0; k < 16; ++k)
      acc = wmma_bf16(afrag[k], frag_linear(bp + k * 512), acc);
    int col = ti * 16 + mr;
    float bias = b_out[col];
#pragma unroll
    for (int r = 0; r < 8; ++r)
      out[(size_t)(row0 + r + 8 * hi) * 512 + col] = acc[r] + bias;
  }
}

extern "C" void kernel_launch(void* const* d_in, const int* in_sizes, int n_in,
                              void* d_out, int out_size, void* d_ws, size_t ws_size,
                              hipStream_t stream) {
  (void)in_sizes; (void)n_in; (void)out_size; (void)ws_size;
  const float* x     = (const float*)d_in[0];
  // d_in[1] = coords (unused: permutation is analytic in the flat index)
  const float* w_qkv = (const float*)d_in[2];
  const float* b_qkv = (const float*)d_in[3];
  const float* gq    = (const float*)d_in[4];
  const float* gk    = (const float*)d_in[5];
  const float* w_out = (const float*)d_in[6];
  const float* b_out = (const float*)d_in[7];
  float* out = (float*)d_out;

  char* ws = (char*)d_ws;
  size_t off = 0;
  float* qkv = (float*)(ws + off); off += (size_t)32768 * 1536 * 4;   // 192 MB
  u16* q_g  = (u16*)(ws + off);    off += (size_t)32768 * 512 * 2;    // 32 MB
  u16* k_p  = (u16*)(ws + off);    off += (size_t)32768 * 512 * 2;    // 32 MB
  u16* v_p  = (u16*)(ws + off);    off += (size_t)32768 * 512 * 2;    // 32 MB
  u16* h_g  = (u16*)(ws + off);    off += (size_t)32768 * 512 * 2;    // 32 MB
  u16* wq_p = (u16*)(ws + off);    off += (size_t)786432 * 2;
  u16* wo_p = (u16*)(ws + off);    off += (size_t)262144 * 2;

  k_pack_w<<<4096, 256, 0, stream>>>(w_qkv, w_out, wq_p, wo_p);
  k_qkv_gemm<<<2048, 256, 0, stream>>>(x, wq_p, b_qkv, qkv);
  k_rms_scatter<<<2048, 256, 0, stream>>>(qkv, gq, gk, q_g, k_p, v_p);
  k_attn<<<2048, 256, 278528, stream>>>(q_g, k_p, v_p, h_g);
  k_out<<<2048, 256, 0, stream>>>(h_g, wo_p, b_out, out);
}
